// DeformableConv2d_42752104465003
// MI455X (gfx1250) — compile-verified, matrix operand order checked by reading, not run on hardware
//
#include <hip/hip_runtime.h>
#include <hip/hip_bf16.h>

typedef __attribute__((ext_vector_type(16))) _Float16     v16h;
typedef __attribute__((ext_vector_type(8)))  float        v8f;
typedef __attribute__((ext_vector_type(4)))  unsigned int u32x4;
typedef __attribute__((ext_vector_type(8)))  unsigned int u32x8;

constexpr int Bn   = 4;
constexpr int Cin  = 64;
constexpr int Hh   = 128;
constexpr int Ww   = 128;
constexpr int Cout = 64;
constexpr int KK   = 9;
constexpr int KDIM = Cin * KK;        // 576
constexpr int HW   = Hh * Ww;         // 16384
constexpr int NPIX = Bn * HW;         // 65536
constexpr int TILE_N = 32;            // pixels per workgroup
constexpr int NBLK = NPIX / TILE_N;   // 2048 workgroups
constexpr int WROW = KDIM;            // weight LDS row stride (halves, unpadded: TDM tile is contiguous)
constexpr int VROW = KDIM + 16;       // Vs LDS row stride (halves, padded)
constexpr int OFFP = 28;              // offset row stride (floats)
constexpr int WREC = 96;              // 64 main + 32 offset weight rows in the f16 blob

// LDS: Wk+Woff [96][WROW] f16, Vs [32][VROW] f16, offs [32][OFFP] f32
constexpr size_t SMEM_BYTES =
    (size_t)WREC * WROW * sizeof(_Float16) +
    (size_t)TILE_N * VROW * sizeof(_Float16) +
    (size_t)TILE_N * OFFP * sizeof(float);

// ---------------------------------------------------------------------------
// Prep: one-time f32 -> f16 weight conversion into ws, laid out [96][576]:
//   rows 0..63  = weight (Cout x C*KK)
//   rows 64..90 = w_off  (27 x C*KK), rows 91..95 = 0 (pad M 27 -> 32)
// ---------------------------------------------------------------------------
__global__ void __launch_bounds__(256)
prep_weights_kernel(const float* __restrict__ weight,
                    const float* __restrict__ w_off,
                    _Float16* __restrict__ wks) {
    int idx = blockIdx.x * 256 + threadIdx.x;
    if (idx < Cout * KDIM) {
        wks[idx] = (_Float16)weight[idx];
    } else if (idx < WREC * KDIM) {
        int j = idx - Cout * KDIM;
        int o = j / KDIM;
        wks[idx] = (o < 27) ? (_Float16)w_off[j] : (_Float16)0.0f;
    }
}

// ISA 7.12.2: 16-bit A/B fragment K-offset for (lane, vgpr v)
__device__ __forceinline__ int frag_k0(int lane, int v) {
    return ((lane & 16) ? 8 : 0) + ((v < 4) ? (2 * v) : (16 + 2 * (v - 4)));
}

// Load one 16x32 (rows x K) fragment from a row-major f16 LDS matrix.
__device__ __forceinline__ v16h load_row_frag(const _Float16* mat, int rowstride,
                                              int row, int kbase, int lane) {
    v16h f;
    const _Float16* r = mat + row * rowstride + kbase;
#pragma unroll
    for (int v = 0; v < 8; ++v) {
        int k0 = frag_k0(lane, v);
        f[2 * v]     = r[k0];
        f[2 * v + 1] = r[k0 + 1];
    }
    return f;
}

__global__ void __launch_bounds__(256)
deform_conv_fused_kernel(const float* __restrict__ x,
                         const float* __restrict__ b_off,
                         const _Float16* __restrict__ wks,
                         const float* __restrict__ bias,
                         float* __restrict__ out) {
    extern __shared__ char smem_raw[];
    _Float16* Wk   = (_Float16*)smem_raw;            // [64][WROW] main weights f16
    _Float16* Woff = Wk + Cout * WROW;               // [32][WROW] offset weights f16
    _Float16* Vs   = Wk + WREC * WROW;               // [32][VROW] im2col, then deformable val
    float*    offs = (float*)(Vs + TILE_N * VROW);   // [32][OFFP] offset-conv outputs

    const int tid  = threadIdx.x;
    const int lane = tid & 31;
    const int wid  = tid >> 5;
    const int p0   = blockIdx.x * TILE_N;

    // ---------------- P0a: TDM DMA of all weights (96x576 f16) ws -> LDS ----------------
    // D# per CDNA5 ISA; 2D tensor, 2-byte elements,
    // tile = whole tensor, contiguous in LDS at Wk. Issued by wave 0 only
    // (EXEC ignored by TDM), overlapped with the im2col build below.
    if (wid == 0) {
        unsigned long long ga = (unsigned long long)(uintptr_t)wks;
        unsigned int ldsa = (unsigned int)(uintptr_t)(void*)Wk;  // flat LDS addr low 32 = LDS offset
        u32x4 g0;
        g0[0] = 1u;                                   // count=1, is_restore=0, no gather
        g0[1] = ldsa;                                 // lds_addr
        g0[2] = (unsigned int)(ga & 0xffffffffu);     // global_addr[31:0]
        g0[3] = (unsigned int)((ga >> 32) & 0x1ffffffu) | (2u << 30); // addr[56:32] | type=2
        u32x8 g1;
        g1[0] = (1u << 16);                           // wg_mask=0, data_size=1 (2 bytes)
        g1[1] = ((unsigned int)KDIM) << 16;           // tensor_dim0[15:0]=576
        g1[2] = ((unsigned int)WREC) << 16;           // tensor_dim0[31:16]=0, tensor_dim1[15:0]=96
        g1[3] = ((unsigned int)KDIM) << 16;           // tensor_dim1[31:16]=0, tile_dim0=576
        g1[4] = (unsigned int)WREC;                   // tile_dim1=96, tile_dim2=0
        g1[5] = (unsigned int)KDIM;                   // tensor_dim0_stride[31:0]=576
        g1[6] = 0u;                                   // stride hi / dim1_stride lo
        g1[7] = 0u;                                   // dim1_stride hi
        asm volatile("tensor_load_to_lds %0, %1"
                     :
                     : "s"(g0), "s"(g1)
                     : "memory");
    }

    // ---------------- P0b: plain im2col of x (for offset conv) ----------------
    for (int t = tid; t < TILE_N * KK; t += 256) {
        int i = t / KK, kk = t - i * KK;
        int ki = kk / 3, kj = kk - ki * 3;
        int p = p0 + i;
        int b = p >> 14, hw = p & (HW - 1);
        int h = hw >> 7, w = hw & (Ww - 1);
        int y = h - 1 + ki, xx = w - 1 + kj;
        bool ok = (y >= 0) && (y < Hh) && (xx >= 0) && (xx < Ww);
        const float* xb = x + (((long)(b * Cin)) << 14) + (y << 7) + xx;
        for (int c = 0; c < Cin; ++c) {
            float v = ok ? xb[c << 14] : 0.0f;
            Vs[i * VROW + c * KK + kk] = (_Float16)v;
        }
    }
    if (wid == 0) __builtin_amdgcn_s_wait_tensorcnt(0);
    __syncthreads();

    // ---------------- P1: offset conv GEMM (32x576 @ 576x32) via WMMA ----------------
    if (wid < 4) {
        const int mt = wid & 1, nt = wid >> 1;
        v8f acc = {};
        for (int kc = 0; kc < KDIM; kc += 32) {
            v16h a  = load_row_frag(Woff, WROW, mt * 16 + (lane & 15), kc, lane);
            v16h bf = load_row_frag(Vs,   VROW, nt * 16 + (lane & 15), kc, lane);
            acc = __builtin_amdgcn_wmma_f32_16x16x32_f16(
                false, a, false, bf, (short)0, acc, false, false);
        }
        int i     = nt * 16 + (lane & 15);
        int obase = mt * 16 + ((lane & 16) ? 8 : 0);
#pragma unroll
        for (int r = 0; r < 8; ++r) {
            int oc = obase + r;
            if (oc < 27) offs[i * OFFP + oc] = acc[r] + b_off[oc];
        }
    }
    __syncthreads();

    // ---------------- P2: deformable bilinear gather -> Vs (f16) ----------------
    for (int t = tid; t < TILE_N * KK; t += 256) {
        int i = t / KK, kk = t - i * KK;
        int ki = kk / 3, kj = kk - ki * 3;
        int p = p0 + i;
        int b = p >> 14, hw = p & (HW - 1);
        int h = hw >> 7, w = hw & (Ww - 1);

        float dy = offs[i * OFFP + 2 * kk];
        float dx = offs[i * OFFP + 2 * kk + 1];
        float mk = offs[i * OFFP + 18 + kk];
        float mask = 1.0f / (1.0f + __expf(-mk));

        float py = (float)(h - 1 + ki) + dy;
        float px = (float)(w - 1 + kj) + dx;
        float fy = floorf(py), fx = floorf(px);
        int y0 = (int)fy, x0 = (int)fx;
        int y1 = y0 + 1,  x1 = x0 + 1;
        float ly = py - fy, lx = px - fx;

        float w00 = (1.f - ly) * (1.f - lx), w01 = (1.f - ly) * lx;
        float w10 = ly * (1.f - lx),         w11 = ly * lx;
        w00 = (y0 >= 0 && y0 < Hh && x0 >= 0 && x0 < Ww) ? w00 : 0.f;
        w01 = (y0 >= 0 && y0 < Hh && x1 >= 0 && x1 < Ww) ? w01 : 0.f;
        w10 = (y1 >= 0 && y1 < Hh && x0 >= 0 && x0 < Ww) ? w10 : 0.f;
        w11 = (y1 >= 0 && y1 < Hh && x1 >= 0 && x1 < Ww) ? w11 : 0.f;

        int yc0 = min(max(y0, 0), Hh - 1), yc1 = min(max(y1, 0), Hh - 1);
        int xc0 = min(max(x0, 0), Ww - 1), xc1 = min(max(x1, 0), Ww - 1);
        int i00 = (yc0 << 7) + xc0, i01 = (yc0 << 7) + xc1;
        int i10 = (yc1 << 7) + xc0, i11 = (yc1 << 7) + xc1;

        const float* xb = x + (((long)(b * Cin)) << 14);
        for (int c = 0; c < Cin; ++c) {
            const float* xc = xb + (c << 14);
            float v = w00 * xc[i00] + w01 * xc[i01] + w10 * xc[i10] + w11 * xc[i11];
            Vs[i * VROW + c * KK + kk] = (_Float16)(v * mask);
        }
    }
    __syncthreads();

    // ---------------- P3: main GEMM (64x576 @ 576x32) via WMMA ----------------
    {
        const int mt = wid & 3, nt = wid >> 2;
        v8f acc = {};
        for (int kc = 0; kc < KDIM; kc += 32) {
            v16h a  = load_row_frag(Wk, WROW, mt * 16 + (lane & 15), kc, lane);
            v16h bf = load_row_frag(Vs, VROW, nt * 16 + (lane & 15), kc, lane);
            acc = __builtin_amdgcn_wmma_f32_16x16x32_f16(
                false, a, false, bf, (short)0, acc, false, false);
        }
        int i = nt * 16 + (lane & 15);
        int p = p0 + i;
        int b = p >> 14, hw = p & (HW - 1);
        int mbase = mt * 16 + ((lane & 16) ? 8 : 0);
        float* ob = out + (((long)(b * Cout)) << 14) + hw;
#pragma unroll
        for (int r = 0; r < 8; ++r) {
            int m = mbase + r;
            ob[(long)m << 14] = acc[r] + bias[m];
        }
    }
}

extern "C" void kernel_launch(void* const* d_in, const int* in_sizes, int n_in,
                              void* d_out, int out_size, void* d_ws, size_t ws_size,
                              hipStream_t stream) {
    const float* x      = (const float*)d_in[0];
    const float* w_off  = (const float*)d_in[1];
    const float* b_off  = (const float*)d_in[2];
    const float* weight = (const float*)d_in[3];
    const float* bias   = (const float*)d_in[4];
    float* out = (float*)d_out;

    _Float16* wks = (_Float16*)d_ws;   // 96*576 f16 = 110,592 bytes

    int prep_elems = WREC * KDIM;
    prep_weights_kernel<<<dim3((prep_elems + 255) / 256), dim3(256), 0, stream>>>(
        weight, w_off, wks);

    deform_conv_fused_kernel<<<dim3(NBLK), dim3(256), SMEM_BYTES, stream>>>(
        x, b_off, wks, bias, out);
}
